// NET_49374944035368
// MI455X (gfx1250) — matrix operand order repaired and checked by reading, not compiled, not run on hardware
//
#include <hip/hip_runtime.h>
#include <hip/hip_bf16.h>

#define EMB   256
#define VOCAB 50000
#define ROWS  101
#define MPAD  112   // 7 * 16

typedef __attribute__((ext_vector_type(16))) _Float16 v16h;
typedef __attribute__((ext_vector_type(8)))  float    v8f;

// ---------------------------------------------------------------------------
// Kernel 0: prep.  W_sum = W_ih + W_hh, b_sum = b_ih + b_hh (valid because the
// decode loop feeds h as both x and h).  Also zero the H padding rows every
// call (workspace is poisoned and never re-poisoned between replays).
// ---------------------------------------------------------------------------
__global__ void prep_kernel(const float* __restrict__ W_ih,
                            const float* __restrict__ W_hh,
                            const float* __restrict__ b_ih,
                            const float* __restrict__ b_hh,
                            float* __restrict__ Wsum,
                            float* __restrict__ bsum,
                            _Float16* __restrict__ Hpad) {
    int i = blockIdx.x * blockDim.x + threadIdx.x;
    if (i < 4 * EMB * EMB) Wsum[i] = W_ih[i] + W_hh[i];
    if (i < 4 * EMB)       bsum[i] = b_ih[i] + b_hh[i];
    if (i < (MPAD - ROWS) * EMB) Hpad[i] = (_Float16)0.f;
}

// ---------------------------------------------------------------------------
// Kernel 1: the serial latency phase (single workgroup, 512 threads).
// conv/bn chain -> W_lin matvec -> 101 sequential LSTM cells, storing each
// hidden state as an f16 row of H[MPAD][EMB].
// ---------------------------------------------------------------------------
__device__ __forceinline__ float sigm(float x) { return 1.f / (1.f + expf(-x)); }

__device__ float conv_bn(const float* xf, float* red, int t,
                         const float* __restrict__ w9,
                         float cbias, float gamma, float beta) {
    float y = 0.f;
    if (t < 361) {
        const int r = t / 19;
        const int c = t - 19 * r;
        #pragma unroll
        for (int dr = 0; dr < 3; ++dr) {
            #pragma unroll
            for (int dc = 0; dc < 3; ++dc) {
                const int rr = r + dr - 1, cc = c + dc - 1;
                if (rr >= 0 && rr < 19 && cc >= 0 && cc < 19)
                    y += w9[dr * 3 + dc] * xf[rr * 19 + cc];
            }
        }
        y += cbias;
    }
    // mean over the 361 elements
    red[t] = (t < 361) ? y : 0.f;
    __syncthreads();
    for (int s = 256; s > 0; s >>= 1) { if (t < s) red[t] += red[t + s]; __syncthreads(); }
    const float mu = red[0] * (1.f / 361.f);
    __syncthreads();
    // biased variance
    const float d = (t < 361) ? (y - mu) : 0.f;
    red[t] = d * d;
    __syncthreads();
    for (int s = 256; s > 0; s >>= 1) { if (t < s) red[t] += red[t + s]; __syncthreads(); }
    const float var = red[0] * (1.f / 361.f);
    __syncthreads();
    return gamma * (y - mu) * rsqrtf(var + 1e-5f) + beta;
}

__global__ __launch_bounds__(512) void serial_phase(
        const float* __restrict__ board,  const float* __restrict__ conv_w,
        const float* __restrict__ conv_b, const float* __restrict__ bn_gamma,
        const float* __restrict__ bn_beta,const float* __restrict__ p,
        const float* __restrict__ W_lin,  const float* __restrict__ b_lin,
        const float* __restrict__ W_ih,   const float* __restrict__ Wsum,
        const float* __restrict__ bsum,   _Float16* __restrict__ H) {
    __shared__ __align__(16) float xf[361];
    __shared__ float red[512];
    __shared__ float gates[4 * EMB];
    __shared__ __align__(16) float xvec[EMB];   // feat, then h
    __shared__ float cbuf[EMB];

    const int t = threadIdx.x;
    const float gamma = bn_gamma[0], beta = bn_beta[0], cbias = conv_b[0];

    if (t < 361) xf[t] = board[t];
    __syncthreads();

    // --- conv/bn chain (fixed structural unroll from the reference) ---
    float y  = conv_bn(xf, red, t, conv_w, cbias, gamma, beta);
    float d1 = fmaxf(p[0] * y, 0.f);
    float d2 = d1;
    #pragma unroll
    for (int ii = 1; ii <= 5; ii += 2) {
        if (t < 361) xf[t] = d1;
        __syncthreads();
        y  = conv_bn(xf, red, t, conv_w, cbias, gamma, beta);
        d1 = fmaxf(p[ii] * y, 0.f);
        if (t < 361) xf[t] = d1;
        __syncthreads();
        y  = conv_bn(xf, red, t, conv_w, cbias, gamma, beta);
        d1 = fmaxf(p[ii + 1] * y + d2, 0.f);
        d2 = d1;
    }
    if (t < 361) xf[t] = d1;
    __syncthreads();
    y  = conv_bn(xf, red, t, conv_w, cbias, gamma, beta);
    d1 = fmaxf(y, 0.f);
    if (t < 361) xf[t] = d1;
    __syncthreads();

    // --- feat = W_lin @ d1 + b_lin ; init LSTM state ---
    if (t < EMB) {
        float acc = b_lin[t];
        const float* wr = W_lin + (size_t)t * 361;
        for (int j = 0; j < 361; ++j) acc += wr[j] * xf[j];
        xvec[t] = acc;      // x input for cell 0 (h0 == 0, so W_hh term vanishes)
        cbuf[t] = 0.f;
    }
    __syncthreads();

    // --- 101 sequential LSTM cells ---
    for (int row = 0; row < ROWS; ++row) {
        // row 0: gates = W_ih @ feat + (b_ih + b_hh)   (h0 = 0)
        // row>0: gates = (W_ih + W_hh) @ h + (b_ih + b_hh)
        const float* Wr = (row == 0) ? W_ih : Wsum;
        float a0 = bsum[t];
        float a1 = bsum[t + 512];
        const float4* w0  = (const float4*)(Wr + (size_t)t * EMB);
        const float4* w1  = (const float4*)(Wr + (size_t)(t + 512) * EMB);
        const float4* xv4 = (const float4*)xvec;
        #pragma unroll 4
        for (int k = 0; k < EMB / 4; ++k) {
            const float4 xq = xv4[k];
            const float4 q0 = w0[k];
            const float4 q1 = w1[k];
            a0 += q0.x * xq.x + q0.y * xq.y + q0.z * xq.z + q0.w * xq.w;
            a1 += q1.x * xq.x + q1.y * xq.y + q1.z * xq.z + q1.w * xq.w;
        }
        gates[t]       = a0;
        gates[t + 512] = a1;
        __syncthreads();

        if (t < EMB) {  // PyTorch gate order i, f, g, o
            const float gi = sigm(gates[t]);
            const float gf = sigm(gates[EMB + t]);
            const float gg = tanhf(gates[2 * EMB + t]);
            const float go = sigm(gates[3 * EMB + t]);
            const float cn = gf * cbuf[t] + gi * gg;
            cbuf[t] = cn;
            const float hn = go * tanhf(cn);
            xvec[t] = hn;
            H[(size_t)row * EMB + t] = (_Float16)hn;
        }
        __syncthreads();
    }
}

// ---------------------------------------------------------------------------
// Kernel 2: out[101,50000] = relu(H @ W_dec^T + b_dec) via v_wmma_f32_16x16x32_f16.
// 3125 blocks x 7 waves; wave w owns M-tile w, block owns a 16-wide N-tile.
// W_dec is fp32 in HBM (read once, L2-resident for reuse), converted to f16 in
// registers per k-step.  A/B fragments share a contiguous K order (k = 16*(lane/16) + s).
// ---------------------------------------------------------------------------
__global__ __launch_bounds__(224) void decode_gemm(
        const _Float16* __restrict__ H,
        const float* __restrict__ Wdec,
        const float* __restrict__ bdec,
        float* __restrict__ out) {
    const int lane  = threadIdx.x & 31;
    const int w     = threadIdx.x >> 5;          // 0..6 -> M-tile
    const int nl    = lane & 15;
    const int ncol  = blockIdx.x * 16 + nl;      // output column
    const int mrowA = 16 * w + nl;               // A-fragment row for this lane
    const int koffA = (lane >> 4) * 16;          // A: lanes 16-31 hold K=16..31
    const int koffB = (lane >> 4) * 16;          // B: lanes 16-31 hold K=16..31

    v8f acc = {};
    #pragma unroll
    for (int kk = 0; kk < EMB / 32; ++kk) {
        const int kb = kk * 32;
        // A fragment: 16 contiguous halves along K of one H row
        const _Float16* Ap = H + (size_t)mrowA * EMB + kb + koffA;
        v16h a;
        #pragma unroll
        for (int i = 0; i < 8; ++i) { a[i] = Ap[i]; a[8 + i] = Ap[8 + i]; }
        // B fragment: 16 contiguous fp32 along K of one W_dec row, cvt->f16
        const float4* Bp = (const float4*)(Wdec + (size_t)ncol * EMB + kb + koffB);
        v16h b;
        #pragma unroll
        for (int i = 0; i < 4; ++i) {
            const float4 q = Bp[i];
            b[4 * i + 0] = (_Float16)q.x;
            b[4 * i + 1] = (_Float16)q.y;
            b[4 * i + 2] = (_Float16)q.z;
            b[4 * i + 3] = (_Float16)q.w;
        }
        acc = __builtin_amdgcn_wmma_f32_16x16x32_f16(
                  false, a, false, b, (short)0, acc, false, false);
    }

    // epilogue: D layout -> VGPR r: lanes 0-15 = row 16w+r, lanes 16-31 = row 16w+8+r
    const float bias  = bdec[ncol];
    const int   mbase = 16 * w + ((lane < 16) ? 0 : 8);
    #pragma unroll
    for (int r = 0; r < 8; ++r) {
        const int m = mbase + r;
        if (m < ROWS) {
            const float v = acc[r] + bias;
            out[(size_t)m * VOCAB + ncol] = v > 0.f ? v : 0.f;
        }
    }
}

// ---------------------------------------------------------------------------
extern "C" void kernel_launch(void* const* d_in, const int* in_sizes, int n_in,
                              void* d_out, int out_size, void* d_ws, size_t ws_size,
                              hipStream_t stream) {
    const float* board    = (const float*)d_in[0];
    const float* conv_w   = (const float*)d_in[1];
    const float* conv_b   = (const float*)d_in[2];
    const float* bn_gamma = (const float*)d_in[3];
    const float* bn_beta  = (const float*)d_in[4];
    const float* p        = (const float*)d_in[5];
    const float* W_lin    = (const float*)d_in[6];
    const float* b_lin    = (const float*)d_in[7];
    const float* W_ih     = (const float*)d_in[8];
    const float* b_ih     = (const float*)d_in[9];
    const float* W_hh     = (const float*)d_in[10];
    const float* b_hh     = (const float*)d_in[11];
    const float* W_dec    = (const float*)d_in[12];
    const float* b_dec    = (const float*)d_in[13];
    float* out = (float*)d_out;

    char* ws = (char*)d_ws;
    float*    Wsum = (float*)ws;                               // 4*256*256*4 = 1 MiB
    float*    bsum = (float*)(ws + 1048576);                   // 4 KiB
    _Float16* H    = (_Float16*)(ws + 1048576 + 4096);         // 112*256*2 = 56 KiB

    prep_kernel<<<dim3((4 * EMB * EMB + 255) / 256), dim3(256), 0, stream>>>(
        W_ih, W_hh, b_ih, b_hh, Wsum, bsum, H + (size_t)ROWS * EMB);

    serial_phase<<<dim3(1), dim3(512), 0, stream>>>(
        board, conv_w, conv_b, bn_gamma, bn_beta, p,
        W_lin, b_lin, W_ih, Wsum, bsum, H);

    decode_gemm<<<dim3(VOCAB / 16), dim3(224), 0, stream>>>(H, W_dec, b_dec, out);
}
